// EGCFModel_42047729828142
// MI455X (gfx1250) — matrix operand order repaired and checked by reading, not compiled, not run on hardware
//
#include <hip/hip_runtime.h>
#include <hip/hip_fp16.h>

typedef __attribute__((ext_vector_type(16))) _Float16 v16h;
typedef __attribute__((ext_vector_type(8)))  float    v8f;

#define U_CNT 100000
#define I_CNT 50000
#define N_CNT 150000
#define K_DIM 64
#define E_CNT 500000
#define NL_CNT 3
#define B_CNT 100000

// ---------------------------------------------------------------- utilities

__global__ void zero_f32(float* __restrict__ p, long long n) {
    long long i = (long long)blockIdx.x * blockDim.x + threadIdx.x;
    long long stride = (long long)gridDim.x * blockDim.x;
    for (; i < n; i += stride) p[i] = 0.0f;
}

// deg[n] = #incident undirected edges (each edge contributes to both endpoints)
__global__ void deg_accum(const int* __restrict__ rows, const int* __restrict__ cols,
                          float* __restrict__ deg) {
    int e = blockIdx.x * blockDim.x + threadIdx.x;
    if (e < E_CNT) {
        unsafeAtomicAdd(&deg[rows[e]], 1.0f);
        unsafeAtomicAdd(&deg[cols[e]], 1.0f);   // cols already offset into [U, U+I)
    }
}

// in-place: deg -> dinv = deg>0 ? rsqrt(deg) : 0
__global__ void dinv_k(float* __restrict__ dd) {
    int n = blockIdx.x * blockDim.x + threadIdx.x;
    if (n < N_CNT) {
        float d = dd[n];
        dd[n] = (d > 0.0f) ? rsqrtf(d) : 0.0f;
    }
}

// normE[e] = dinv[rows[e]] * dinv[cols[e]]  (symmetric in edge direction)
__global__ void norm_k(const int* __restrict__ rows, const int* __restrict__ cols,
                       const float* __restrict__ dinv, float* __restrict__ normE) {
    int e = blockIdx.x * blockDim.x + threadIdx.x;
    if (e < E_CNT) normE[e] = dinv[rows[e]] * dinv[cols[e]];
}

// C0 = concat(Gu, Gi), T0 = concat(Gut, Git)
__global__ void init_k(const float* __restrict__ Gu, const float* __restrict__ Gi,
                       const float* __restrict__ Gut, const float* __restrict__ Git,
                       float* __restrict__ C0, float* __restrict__ T0) {
    long long i = (long long)blockIdx.x * blockDim.x + threadIdx.x;
    if (i >= (long long)N_CNT * K_DIM) return;
    long long n = i >> 6;
    int k = (int)(i & 63);
    if (n < U_CNT) {
        C0[i] = Gu[n * K_DIM + k];
        T0[i] = Gut[n * K_DIM + k];
    } else {
        long long m = n - U_CNT;
        C0[i] = Gi[m * K_DIM + k];
        T0[i] = Git[m * K_DIM + k];
    }
}

// ---------------------------------------------------------------- WMMA GEMM
// ep[E x 64] = ef[E x 64] @ W[64 x 64] + bias
// Per wave: 16-edge x 64-col tile; K=64 split into two 16x16x32 f16 WMMA steps
// per N-tile (4 N-tiles) = 8 v_wmma per tile. B fragments built once per wave
// (W is 16KB, cache-resident) and reused across a grid-stride tile loop.

__global__ void __launch_bounds__(256) ep_gemm_wmma(const float* __restrict__ ef,
                                                    const float* __restrict__ W,
                                                    const float* __restrict__ bias,
                                                    float* __restrict__ ep) {
    const int lane = threadIdx.x & 31;
    const int wave = threadIdx.x >> 5;
    const int ln15 = lane & 15;
    const int hi   = (lane >> 4) & 1;       // lane half
    const int kb16 = hi * 16;               // unused; B also uses kb8
    const int kb8  = hi * 8;                // A: K offset for this lane half

    // --- B fragments: element i of v16h holds W[fbase + kb8 + i + (i>=8 ? 8 : 0)][n], n = t*16+ln15
    v16h Bf[2][4];
    float bb[4];
#pragma unroll
    for (int s = 0; s < 2; ++s) {
#pragma unroll
        for (int t = 0; t < 4; ++t) {
            const int n = t * 16 + ln15;
#pragma unroll
            for (int i = 0; i < 16; ++i) {
                const int f = s * 32 + kb8 + i + (i >= 8 ? 8 : 0);
                Bf[s][t][i] = (_Float16)W[f * 64 + n];
            }
        }
    }
#pragma unroll
    for (int t = 0; t < 4; ++t) bb[t] = bias[t * 16 + ln15];

    const long long numTiles = E_CNT / 16;  // E % 16 == 0, no remainder
    for (long long tile = (long long)blockIdx.x * 8 + wave; tile < numTiles;
         tile += (long long)gridDim.x * 8) {
        const long long row0 = tile * 16;
        const float* ar = ef + (row0 + ln15) * 64;

        // --- A fragments: 16x32 f16 layout. Element i (0..7) -> k = kb8+i,
        //     element i (8..15) -> k = 16+kb8+(i-8); fstep s adds 32.
        v16h Afr[2];
#pragma unroll
        for (int s = 0; s < 2; ++s) {
            const float4 c0 = *(const float4*)(ar + s * 32 + kb8);
            const float4 c1 = *(const float4*)(ar + s * 32 + kb8 + 4);
            const float4 c2 = *(const float4*)(ar + s * 32 + kb8 + 16);
            const float4 c3 = *(const float4*)(ar + s * 32 + kb8 + 20);
            Afr[s][0]  = (_Float16)c0.x; Afr[s][1]  = (_Float16)c0.y;
            Afr[s][2]  = (_Float16)c0.z; Afr[s][3]  = (_Float16)c0.w;
            Afr[s][4]  = (_Float16)c1.x; Afr[s][5]  = (_Float16)c1.y;
            Afr[s][6]  = (_Float16)c1.z; Afr[s][7]  = (_Float16)c1.w;
            Afr[s][8]  = (_Float16)c2.x; Afr[s][9]  = (_Float16)c2.y;
            Afr[s][10] = (_Float16)c2.z; Afr[s][11] = (_Float16)c2.w;
            Afr[s][12] = (_Float16)c3.x; Afr[s][13] = (_Float16)c3.y;
            Afr[s][14] = (_Float16)c3.z; Afr[s][15] = (_Float16)c3.w;
        }

        v8f acc[4] = {};
#pragma unroll
        for (int t = 0; t < 4; ++t) {
            acc[t] = __builtin_amdgcn_wmma_f32_16x16x32_f16(
                false, Afr[0], false, Bf[0][t], (short)0, acc[t], false, false);
            acc[t] = __builtin_amdgcn_wmma_f32_16x16x32_f16(
                false, Afr[1], false, Bf[1][t], (short)0, acc[t], false, false);
        }

        // --- store: C/D layout: VGPR r -> row M = r + 8*hi, col N = t*16+ln15
        float* orow = ep + row0 * 64;
#pragma unroll
        for (int t = 0; t < 4; ++t) {
            const int col = t * 16 + ln15;
#pragma unroll
            for (int r = 0; r < 8; ++r) {
                const int m = r + hi * 8;
                orow[(long long)m * 64 + col] = acc[t][r] + bb[t];
            }
        }
    }
}

// ---------------------------------------------------------------- propagation
// One directed edge per (blockIdx.x, threadIdx.y); threadIdx.x = k (coalesced).
__global__ void __launch_bounds__(256) prop_layer(const int* __restrict__ rows,
                                                  const int* __restrict__ cols,
                                                  const float* __restrict__ normE,
                                                  const float* __restrict__ ep,
                                                  const float* __restrict__ curC,
                                                  const float* __restrict__ curT,
                                                  float* __restrict__ nxtC,
                                                  float* __restrict__ nxtT) {
    long long de = (long long)blockIdx.x * blockDim.y + threadIdx.y;
    if (de >= 2LL * E_CNT) return;
    const int k = threadIdx.x;
    int em, s, d;
    if (de < E_CNT) {
        em = (int)de;        s = rows[em]; d = cols[em];
    } else {
        em = (int)(de - E_CNT); s = cols[em]; d = rows[em];
    }
    const float w = normE[em];
    const float cv = curC[(long long)s * 64 + k] * w;
    const float tv = curT[(long long)s * 64 + k] * ep[(long long)em * 64 + k] * w;
    unsafeAtomicAdd(&nxtC[(long long)d * 64 + k], cv);
    unsafeAtomicAdd(&nxtT[(long long)d * 64 + k], tv);
}

// ---------------------------------------------------------------- scoring
__global__ void score_k(const float* __restrict__ C, const float* __restrict__ T,
                        const float* __restrict__ Bu, const float* __restrict__ Bi,
                        const float* __restrict__ But, const float* __restrict__ Bit,
                        const float* __restrict__ Mu, const int* __restrict__ users,
                        const int* __restrict__ items, float* __restrict__ out) {
    int b = blockIdx.x * blockDim.x + threadIdx.x;
    if (b >= B_CNT) return;
    const int u  = users[b];
    const int it = items[b];
    const float4* gu  = (const float4*)(C + (long long)u * 64);
    const float4* gi  = (const float4*)(C + ((long long)U_CNT + it) * 64);
    const float4* gut = (const float4*)(T + (long long)u * 64);
    const float4* git = (const float4*)(T + ((long long)U_CNT + it) * 64);
    float acc = 0.0f;
#pragma unroll
    for (int q = 0; q < 16; ++q) {
        float4 a = gu[q],  c = gi[q];
        acc += a.x * c.x + a.y * c.y + a.z * c.z + a.w * c.w;
        float4 e = gut[q], f = git[q];
        acc += e.x * f.x + e.y * f.y + e.z * f.z + e.w * f.w;
    }
    out[b] = acc + Bu[u] + Bi[it] + But[u] + Bit[it] + Mu[0];
}

// ---------------------------------------------------------------- launch

extern "C" void kernel_launch(void* const* d_in, const int* in_sizes, int n_in,
                              void* d_out, int out_size, void* d_ws, size_t ws_size,
                              hipStream_t stream) {
    const float* Gu  = (const float*)d_in[0];
    const float* Gi  = (const float*)d_in[1];
    const float* Gut = (const float*)d_in[2];
    const float* Git = (const float*)d_in[3];
    const float* Bu  = (const float*)d_in[4];
    const float* Bi  = (const float*)d_in[5];
    const float* But = (const float*)d_in[6];
    const float* Bit = (const float*)d_in[7];
    const float* Mu  = (const float*)d_in[8];
    const float* Wp  = (const float*)d_in[9];
    const float* bp  = (const float*)d_in[10];
    const float* ef  = (const float*)d_in[11];
    const int* rows  = (const int*)d_in[12];
    const int* cols  = (const int*)d_in[13];
    const int* users = (const int*)d_in[14];
    const int* items = (const int*)d_in[15];
    float* out = (float*)d_out;

    // workspace layout (floats); total ~284 MB
    float* ws = (float*)d_ws;
    size_t off = 0;
    float* dinv  = ws + off; off += ((N_CNT + 63) & ~63ULL);   // deg then dinv
    float* normE = ws + off; off += E_CNT;
    float* ep    = ws + off; off += (size_t)E_CNT * K_DIM;
    float* C0    = ws + off; off += (size_t)N_CNT * K_DIM;
    float* C1    = ws + off; off += (size_t)N_CNT * K_DIM;
    float* T0    = ws + off; off += (size_t)N_CNT * K_DIM;
    float* T1    = ws + off; off += (size_t)N_CNT * K_DIM;
    (void)ws_size; (void)in_sizes; (void)n_in; (void)out_size;

    zero_f32<<<512, 256, 0, stream>>>(dinv, N_CNT);
    deg_accum<<<(E_CNT + 255) / 256, 256, 0, stream>>>(rows, cols, dinv);
    dinv_k<<<(N_CNT + 255) / 256, 256, 0, stream>>>(dinv);
    norm_k<<<(E_CNT + 255) / 256, 256, 0, stream>>>(rows, cols, dinv, normE);
    init_k<<<(int)(((long long)N_CNT * K_DIM + 255) / 256), 256, 0, stream>>>(
        Gu, Gi, Gut, Git, C0, T0);
    ep_gemm_wmma<<<1536, 256, 0, stream>>>(ef, Wp, bp, ep);

    float* curC = C0; float* curT = T0;
    float* nxtC = C1; float* nxtT = T1;
    for (int l = 0; l < NL_CNT; ++l) {
        zero_f32<<<2048, 256, 0, stream>>>(nxtC, (long long)N_CNT * K_DIM);
        zero_f32<<<2048, 256, 0, stream>>>(nxtT, (long long)N_CNT * K_DIM);
        dim3 blk(64, 4);
        prop_layer<<<(2 * E_CNT + 3) / 4, blk, 0, stream>>>(
            rows, cols, normE, ep, curC, curT, nxtC, nxtT);
        float* t;
        t = curC; curC = nxtC; nxtC = t;
        t = curT; curT = nxtT; nxtT = t;
    }

    score_k<<<(B_CNT + 255) / 256, 256, 0, stream>>>(
        curC, curT, Bu, Bi, But, Bit, Mu, users, items, out);
}
